// FusedEPMoE_17136919511770
// MI455X (gfx1250) — compile-verified, matrix operand order checked by reading, not run on hardware
//
#include <hip/hip_runtime.h>
#include <hip/hip_bf16.h>
#include <math.h>

// ---------------------------------------------------------------------------
// Fused MoE (SwiGLU, top-2 of 8 experts) for gfx1250 / MI455X.
// fp32 inputs; GEMMs run as bf16-split (hi/lo) v_wmma_f32_16x16x32_bf16 with
// fp32 accumulation ("bf16x3": ah*bh + ah*bl + al*bh  ~ fp32 accuracy).
// The hi/lo split is done once at LDS-staging time; LDS holds bf16 planes so
// WMMA fragments are two aligned ds_load_b128 each (no VALU assembly).
// ---------------------------------------------------------------------------

static constexpr int kT = 2048;   // tokens
static constexpr int kH = 1024;   // hidden
static constexpr int kI = 1024;   // intermediate
static constexpr int kE = 8;      // experts
static constexpr int kK = 2;      // top-k

static constexpr int BM = 32;       // token rows per workgroup tile
static constexpr int BN = 64;       // output cols per workgroup tile
static constexpr int KS = 32;       // k-depth per step (bf16 wmma K)
static constexpr int MT = kT / BM;  // max m-tiles per expert
static constexpr int LDP2 = 40;     // LDS row stride in bf16 (80 B: 16B-aligned, staggers banks)

typedef __attribute__((ext_vector_type(16))) __bf16 v16bf;
typedef __attribute__((ext_vector_type(8)))  __bf16 v8bf;
typedef __attribute__((ext_vector_type(8)))  float  v8f;

__device__ __forceinline__ v8f wmma_bf16(v16bf a, v16bf b, v8f c) {
  // (neg_a, A, neg_b, B, c_mod, C, reuse_a, reuse_b)
  return __builtin_amdgcn_wmma_f32_16x16x32_bf16(false, a, false, b,
                                                 (short)0, c, false, false);
}

// 16-bit A/B fragment (16x32) from a bf16 LDS plane laid out [row][k]:
// lane L: row = L&15, koff = 8*(L>>4); dwords 0..3 hold k in [koff,koff+8),
// dwords 4..7 hold k in [16+koff,16+koff+8)  ->  two aligned ds_load_b128.
__device__ __forceinline__ v16bf load_frag(const __bf16* lds, int row, int koff) {
  const __bf16* p = lds + row * LDP2;
  v8bf a = *(const v8bf*)(p + koff);
  v8bf b = *(const v8bf*)(p + 16 + koff);
  return __builtin_shufflevector(a, b, 0, 1, 2, 3, 4, 5, 6, 7,
                                 8, 9, 10, 11, 12, 13, 14, 15);
}

union BfPack4 { __bf16 b[4]; uint2 u; };

// split 4 fp32 into hi/lo bf16 quads
__device__ __forceinline__ void split4(const float4 v, BfPack4& hi, BfPack4& lo) {
  float f[4] = {v.x, v.y, v.z, v.w};
#pragma unroll
  for (int j = 0; j < 4; ++j) {
    __bf16 h = (__bf16)f[j];
    hi.b[j] = h;
    lo.b[j] = (__bf16)(f[j] - (float)h);
  }
}

// ---------------------------------------------------------------------------
// 1) Routing: softmax over E=8 logits, top-2 (ties -> lowest index, matching
//    jax.lax.top_k), bucket tokens per expert. Slot order may vary run-to-run
//    but the result is slot-order independent (see combine).
// ---------------------------------------------------------------------------
__global__ __launch_bounds__(256) void routing_kernel(
    const float* __restrict__ logits, int* __restrict__ cnt,
    int* __restrict__ tok_of, int* __restrict__ tk_e,
    int* __restrict__ tk_pos, float* __restrict__ tk_w) {
  int t = blockIdx.x * blockDim.x + threadIdx.x;
  if (t >= kT) return;
  float p[kE];
  float mx = -INFINITY;
#pragma unroll
  for (int e = 0; e < kE; ++e) { p[e] = logits[t * kE + e]; mx = fmaxf(mx, p[e]); }
  float s = 0.f;
#pragma unroll
  for (int e = 0; e < kE; ++e) { p[e] = expf(p[e] - mx); s += p[e]; }
  float inv = 1.f / s;
#pragma unroll
  for (int e = 0; e < kE; ++e) p[e] *= inv;

  int e1 = 0;
#pragma unroll
  for (int e = 1; e < kE; ++e) if (p[e] > p[e1]) e1 = e;
  int e2 = (e1 == 0) ? 1 : 0;
#pragma unroll
  for (int e = 0; e < kE; ++e) if (e != e1 && e != e2 && p[e] > p[e2]) e2 = e;

  int   sel_e[kK] = {e1, e2};
  float sel_w[kK] = {p[e1], p[e2]};
#pragma unroll
  for (int k = 0; k < kK; ++k) {
    int e = sel_e[k];
    int pos = atomicAdd(&cnt[e], 1);
    tok_of[e * kT + pos] = t;
    tk_e[t * kK + k] = e;
    tk_pos[t * kK + k] = pos;
    tk_w[t * kK + k] = sel_w[k];
  }
}

__global__ void prefix_kernel(const int* __restrict__ cnt, int* __restrict__ base) {
  if (threadIdx.x == 0) {
    int acc = 0;
#pragma unroll
    for (int e = 0; e < kE; ++e) { base[e] = acc; acc += cnt[e]; }
  }
}

// ---------------------------------------------------------------------------
// 2) Grouped GEMM1 + SwiGLU:  h[row, n] = silu(x@w1) * (x@w3)
//    rows = gathered tokens of expert e (compact rows base[e]+slot).
//    Writes h as bf16 hi/lo planes so GEMM2 staging is conversion-free.
// ---------------------------------------------------------------------------
__global__ __launch_bounds__(256) void gemm1_swiglu_kernel(
    const float* __restrict__ X, const float* __restrict__ W1,
    const float* __restrict__ W3, const int* __restrict__ cnt,
    const int* __restrict__ base, const int* __restrict__ tok_of,
    __bf16* __restrict__ hbh, __bf16* __restrict__ hbl) {
  const int e  = blockIdx.x / MT;
  const int mt = blockIdx.x % MT;
  const int ce = cnt[e];
  if (mt * BM >= ce) return;
  const int n0 = blockIdx.y * BN;

  __shared__ __align__(16) __bf16 sxh[BM * LDP2];
  __shared__ __align__(16) __bf16 sxl[BM * LDP2];
  __shared__ __align__(16) __bf16 s1h[BN * LDP2];
  __shared__ __align__(16) __bf16 s1l[BN * LDP2];
  __shared__ __align__(16) __bf16 s3h[BN * LDP2];
  __shared__ __align__(16) __bf16 s3l[BN * LDP2];
  __shared__ int tokrow[BM];

  const int tid = threadIdx.x;
  if (tid < BM) {
    int r = mt * BM + tid;
    tokrow[tid] = tok_of[e * kT + ((r < ce) ? r : (ce - 1))];
  }
  __syncthreads();

  const int lane = tid & 31;
  const int wave = tid >> 5;
  const int wm = wave >> 2;            // 0..1
  const int wn = wave & 3;             // 0..3
  const int mn   = lane & 15;
  const int koff = (lane >> 4) * 8;

  const float* W1e = W1 + (size_t)e * kH * kI;
  const float* W3e = W3 + (size_t)e * kH * kI;

  v8f cg = {};
  v8f cu = {};

  for (int k0 = 0; k0 < kH; k0 += KS) {
    {   // stage X tile (BM x KS) split to bf16 hi/lo planes
      int r  = tid >> 3;
      int kq = (tid & 7) * 4;
      const float4 xv = *(const float4*)(X + (size_t)tokrow[r] * kH + k0 + kq);
      BfPack4 h, l;
      split4(xv, h, l);
      *(uint2*)&sxh[r * LDP2 + kq] = h.u;
      *(uint2*)&sxl[r * LDP2 + kq] = l.u;
    }
    {   // stage W1/W3 tiles transposed to [n][k], split to bf16 hi/lo planes
      int kk = tid >> 4;               // 0..15
      int nq = (tid & 15) * 4;         // 0..60
#pragma unroll
      for (int khalf = 0; khalf < 2; ++khalf) {
        int k = kk + khalf * 16;
        const float4 a = *(const float4*)(W1e + (size_t)(k0 + k) * kI + n0 + nq);
        const float4 b = *(const float4*)(W3e + (size_t)(k0 + k) * kI + n0 + nq);
        BfPack4 ah, al, bh, bl;
        split4(a, ah, al);
        split4(b, bh, bl);
#pragma unroll
        for (int j = 0; j < 4; ++j) {
          s1h[(nq + j) * LDP2 + k] = ah.b[j];
          s1l[(nq + j) * LDP2 + k] = al.b[j];
          s3h[(nq + j) * LDP2 + k] = bh.b[j];
          s3l[(nq + j) * LDP2 + k] = bl.b[j];
        }
      }
      if (k0 + KS < kH) {              // hint next weight k-tile toward L2
        __builtin_prefetch(W1e + (size_t)(k0 + KS + kk) * kI + n0, 0, 1);
        __builtin_prefetch(W3e + (size_t)(k0 + KS + kk) * kI + n0, 0, 1);
      }
    }
    __syncthreads();

    v16bf axh = load_frag(sxh, wm * 16 + mn, koff);
    v16bf axl = load_frag(sxl, wm * 16 + mn, koff);
    v16bf b1h = load_frag(s1h, wn * 16 + mn, koff);
    v16bf b1l = load_frag(s1l, wn * 16 + mn, koff);
    v16bf b3h = load_frag(s3h, wn * 16 + mn, koff);
    v16bf b3l = load_frag(s3l, wn * 16 + mn, koff);

    cg = wmma_bf16(axh, b1h, cg);
    cg = wmma_bf16(axh, b1l, cg);
    cg = wmma_bf16(axl, b1h, cg);
    cu = wmma_bf16(axh, b3h, cu);
    cu = wmma_bf16(axh, b3l, cu);
    cu = wmma_bf16(axl, b3h, cu);
    __syncthreads();
  }

  // epilogue: SwiGLU, write compact h rows as bf16 hi/lo planes
  const int ncol  = n0 + wn * 16 + (lane & 15);
  const int rbase = mt * BM + wm * 16 + ((lane >> 4) * 8);
  const int be = base[e];
#pragma unroll
  for (int r = 0; r < 8; ++r) {
    int row = rbase + r;
    if (row < ce) {
      float g = cg[r];
      float u = cu[r];
      float h = g / (1.0f + expf(-g)) * u;
      __bf16 hh = (__bf16)h;
      size_t idx = (size_t)(be + row) * kI + ncol;
      hbh[idx] = hh;
      hbl[idx] = (__bf16)(h - (float)hh);
    }
  }
}

// ---------------------------------------------------------------------------
// 3) Grouped GEMM2:  contrib[row, n] = h[row, :] @ w2[e]   (unscaled, so the
//    combine step is deterministic; no fp atomics anywhere).
//    A tiles are already bf16 hi/lo -> staging is a pure copy.
// ---------------------------------------------------------------------------
__global__ __launch_bounds__(256) void gemm2_kernel(
    const __bf16* __restrict__ hbh, const __bf16* __restrict__ hbl,
    const float* __restrict__ W2, const int* __restrict__ cnt,
    const int* __restrict__ base, float* __restrict__ contrib) {
  const int e  = blockIdx.x / MT;
  const int mt = blockIdx.x % MT;
  const int ce = cnt[e];
  if (mt * BM >= ce) return;
  const int n0 = blockIdx.y * BN;
  const int be = base[e];

  __shared__ __align__(16) __bf16 shh[BM * LDP2];
  __shared__ __align__(16) __bf16 shl[BM * LDP2];
  __shared__ __align__(16) __bf16 swh[BN * LDP2];
  __shared__ __align__(16) __bf16 swl[BN * LDP2];

  const int tid = threadIdx.x;
  const int lane = tid & 31;
  const int wave = tid >> 5;
  const int wm = wave >> 2;
  const int wn = wave & 3;
  const int mn   = lane & 15;
  const int koff = (lane >> 4) * 8;

  const float* W2e = W2 + (size_t)e * kI * kH;

  v8f acc = {};

  for (int k0 = 0; k0 < kI; k0 += KS) {
    {   // stage h tile: straight bf16 copy (4 bf16 = 8 B per thread per plane)
      int r  = tid >> 3;
      int kq = (tid & 7) * 4;
      int row = mt * BM + r;
      int rowg = be + ((row < ce) ? row : (ce - 1));
      size_t gidx = (size_t)rowg * kI + k0 + kq;
      *(uint2*)&shh[r * LDP2 + kq] = *(const uint2*)(hbh + gidx);
      *(uint2*)&shl[r * LDP2 + kq] = *(const uint2*)(hbl + gidx);
    }
    {   // stage W2 tile transposed to [n][k], split to bf16 hi/lo planes
      int kk = tid >> 4;
      int nq = (tid & 15) * 4;
#pragma unroll
      for (int khalf = 0; khalf < 2; ++khalf) {
        int k = kk + khalf * 16;
        const float4 a = *(const float4*)(W2e + (size_t)(k0 + k) * kH + n0 + nq);
        BfPack4 ah, al;
        split4(a, ah, al);
#pragma unroll
        for (int j = 0; j < 4; ++j) {
          swh[(nq + j) * LDP2 + k] = ah.b[j];
          swl[(nq + j) * LDP2 + k] = al.b[j];
        }
      }
      if (k0 + KS < kI)
        __builtin_prefetch(W2e + (size_t)(k0 + KS + kk) * kH + n0, 0, 1);
    }
    __syncthreads();

    v16bf ah = load_frag(shh, wm * 16 + mn, koff);
    v16bf al = load_frag(shl, wm * 16 + mn, koff);
    v16bf bh = load_frag(swh, wn * 16 + mn, koff);
    v16bf bl = load_frag(swl, wn * 16 + mn, koff);
    acc = wmma_bf16(ah, bh, acc);
    acc = wmma_bf16(ah, bl, acc);
    acc = wmma_bf16(al, bh, acc);
    __syncthreads();
  }

  const int ncol  = n0 + wn * 16 + (lane & 15);
  const int rbase = mt * BM + wm * 16 + ((lane >> 4) * 8);
#pragma unroll
  for (int r = 0; r < 8; ++r) {
    int row = rbase + r;
    if (row < ce)
      contrib[(size_t)(be + row) * kH + ncol] = acc[r];
  }
}

// ---------------------------------------------------------------------------
// 4) Combine: out[t, col] = sum_k w[t,k] * contrib[base[e_k]+pos_k, col]
// ---------------------------------------------------------------------------
__global__ __launch_bounds__(256) void combine_kernel(
    const float* __restrict__ contrib, const int* __restrict__ base,
    const int* __restrict__ tk_e, const int* __restrict__ tk_pos,
    const float* __restrict__ tk_w, float* __restrict__ out) {
  int idx = blockIdx.x * blockDim.x + threadIdx.x;
  if (idx >= kT * kH) return;
  int t   = idx >> 10;        // / kH
  int col = idx & (kH - 1);
  float acc = 0.f;
#pragma unroll
  for (int k = 0; k < kK; ++k) {
    int e   = tk_e[t * kK + k];
    int pos = tk_pos[t * kK + k];
    float w = tk_w[t * kK + k];
    acc += w * contrib[(size_t)(base[e] + pos) * kH + col];
  }
  out[idx] = acc;
}

// ---------------------------------------------------------------------------
// Launch
// ---------------------------------------------------------------------------
extern "C" void kernel_launch(void* const* d_in, const int* in_sizes, int n_in,
                              void* d_out, int out_size, void* d_ws, size_t ws_size,
                              hipStream_t stream) {
  (void)in_sizes; (void)n_in; (void)out_size; (void)ws_size;
  const float* X      = (const float*)d_in[0];   // (T,H)
  const float* logits = (const float*)d_in[1];   // (T,E)
  const float* w1     = (const float*)d_in[2];   // (E,H,I)
  const float* w2     = (const float*)d_in[3];   // (E,I,H)
  const float* w3     = (const float*)d_in[4];   // (E,H,I)
  float* out = (float*)d_out;

  // Workspace layout (bytes); total ~34 MB.
  char* ws = (char*)d_ws;
  int*    cnt     = (int*)(ws + 0);                       // 8 ints
  int*    base    = (int*)(ws + 256);                     // 8 ints
  int*    tok_of  = (int*)(ws + 512);                     // E*T ints   (64 KB)
  int*    tk_e    = (int*)(ws + 512 + 65536);             // T*K ints   (16 KB)
  int*    tk_pos  = (int*)(ws + 512 + 65536 + 16384);     // T*K ints
  float*  tk_w    = (float*)(ws + 512 + 65536 + 32768);   // T*K floats
  __bf16* hbh     = (__bf16*)(ws + (1 << 20));                               // 4096*1024 bf16 (8 MB)
  __bf16* hbl     = (__bf16*)(ws + (1 << 20) + (size_t)8388608);             // 8 MB
  float*  contrib = (float*)(ws + (1 << 20) + (size_t)16777216);             // 16 MB

  hipMemsetAsync(cnt, 0, 64, stream);

  routing_kernel<<<kT / 256, 256, 0, stream>>>(logits, cnt, tok_of, tk_e, tk_pos, tk_w);
  prefix_kernel<<<1, 32, 0, stream>>>(cnt, base);

  dim3 g1(kE * MT, kI / BN);
  gemm1_swiglu_kernel<<<g1, 256, 0, stream>>>(X, w1, w3, cnt, base, tok_of, hbh, hbl);

  dim3 g2(kE * MT, kH / BN);
  gemm2_kernel<<<g2, 256, 0, stream>>>(hbh, hbl, w2, cnt, base, contrib);

  combine_kernel<<<(kT * kH) / 256, 256, 0, stream>>>(contrib, base, tk_e, tk_pos, tk_w, out);
}
